// CAGpool_50981261804245
// MI455X (gfx1250) — matrix operand order repaired and checked by reading, not compiled, not run on hardware
//
#include <hip/hip_runtime.h>
#include <math.h>

// ---- problem constants (from reference) ----
#define BQ      64
#define N1Q     512            // nodes per batch per subgraph
#define NPB     1024           // nodes per batch total (N1+N2)
#define NTOT    65536          // B*(N1+N2)
#define NFQ     128
#define NHQ     128
#define NCATQ   384            // NH*NL
#define KSEL    256            // top-k per batch
#define E_ALL   1048576        // edges in combined graph
#define E_C     524288         // edges per subgraph

typedef __attribute__((ext_vector_type(16))) __bf16 v16bf;
typedef __attribute__((ext_vector_type(8)))  float  v8f;

enum { ACT_NONE = 0, ACT_RELU = 1, ACT_TANH = 2 };

__device__ __forceinline__ float sigmoidf_(float x) { return 1.0f / (1.0f + expf(-x)); }

// ---------------------------------------------------------------------------
// Generic GEMM: C = act( A' @ W + bias ), A'[m,:] = A[rowmap?rowmap[m]:m,:]*rowscale[m]
// A: M x K (row stride lda, f32), W: K x Nc (row stride ldw, f32), C row stride ldc.
// One wave computes a 16x16 tile via v_wmma_f32_16x16x32_bf16, K unrolled x2
// (all call sites have K % 64 == 0) so two WMMAs issue per iteration and the
// bf16-convert VALU work co-executes under the matrix op (ISA 8.4: XDL WMMA
// tracked as TRANS -> co-execution with VALU).
// Fragment layouts per CDNA5 ISA 7.12.2 (wave32):
//   A 16x32 bf16 : lanes (h, m=lane&15) hold row m; elem e -> K = (e<8 ? h*8+e
//                  : 16 + h*8 + e-8)  (two contiguous 8-float runs per lane)
//   B 32x16 bf16 : lane (h, n=lane&15): elem e -> K = (e<8 ? h*8+e : 16 + h*8 + e-8), col n
//   C 16x16 f32  : vgpr r -> row = h*8+r, col = lane&15
// ---------------------------------------------------------------------------
__device__ __forceinline__ v16bf load_a_frag(const float* __restrict__ Ap,
                                             int h, float sc)
{
  v16bf a;
#pragma unroll
  for (int e = 0; e < 8; ++e) {
    a[e]     = (__bf16)(Ap[h * 8 + e] * sc);
    a[e + 8] = (__bf16)(Ap[16 + h * 8 + e] * sc);
  }
  return a;
}

__device__ __forceinline__ v16bf load_b_frag(const float* __restrict__ Wp,
                                             int h, int ldw)
{
  v16bf b;
#pragma unroll
  for (int e = 0; e < 8; ++e) {
    b[e]     = (__bf16)(Wp[(long)(h * 8 + e) * ldw]);
    b[e + 8] = (__bf16)(Wp[(long)(16 + h * 8 + e) * ldw]);
  }
  return b;
}

__global__ void gemm_wmma(const float* __restrict__ A, int lda,
                          const float* __restrict__ W, int ldw,
                          float* __restrict__ C, int ldc,
                          const float* __restrict__ bias,
                          const int*   __restrict__ rowmap,
                          const float* __restrict__ rowscale,
                          int M, int K, int Nc, int act)
{
  const int ntiles = Nc >> 4;
  const int wave = (int)((blockIdx.x * (long)blockDim.x + threadIdx.x) >> 5);
  const int lane = threadIdx.x & 31;
  const int mt = wave / ntiles;
  const int nt = wave - mt * ntiles;
  if (mt * 16 >= M) return;              // wave-uniform exit: EXEC stays all-ones
  const int h  = lane >> 4;
  const int ml = lane & 15;

  const int   arow = mt * 16 + ml;
  const int   srow = rowmap ? rowmap[arow] : arow;
  const float sc   = rowscale ? rowscale[arow] : 1.0f;
  const float* Abase = A + (long)srow * lda;
  const float* Wbase = W + nt * 16 + ml;

  v8f acc = {};
  for (int kk = 0; kk < K; kk += 64) {     // two K=32 slices per iteration
    v16bf a0 = load_a_frag(Abase + kk, h, sc);
    v16bf b0 = load_b_frag(Wbase + (long)kk * ldw, h, ldw);
    v16bf a1 = load_a_frag(Abase + kk + 32, h, sc);
    v16bf b1 = load_b_frag(Wbase + (long)(kk + 32) * ldw, h, ldw);
    acc = __builtin_amdgcn_wmma_f32_16x16x32_bf16(false, a0, false, b0,
                                                  (short)0, acc, false, false);
    acc = __builtin_amdgcn_wmma_f32_16x16x32_bf16(false, a1, false, b1,
                                                  (short)0, acc, false, false);
  }

  const int col = nt * 16 + ml;
  const float bv = bias ? bias[col] : 0.0f;
#pragma unroll
  for (int r = 0; r < 8; ++r) {
    float vv = acc[r] + bv;
    if (act == ACT_RELU)      vv = fmaxf(vv, 0.0f);
    else if (act == ACT_TANH) vv = tanhf(vv);
    C[(long)(mt * 16 + h * 8 + r) * ldc + col] = vv;
  }
}

// ---------------------------------------------------------------------------
// Small utility kernels
// ---------------------------------------------------------------------------
__global__ void fill_f32(float* p, float v, int n)
{
  int i = blockIdx.x * blockDim.x + threadIdx.x;
  if (i < n) p[i] = v;
}

// deg[dst] += 1 for unmasked edges (nmap: optional remap; edge masked if either end < 0)
__global__ void deg_accum(const int* __restrict__ src, const int* __restrict__ dst,
                          const int* __restrict__ nmap, float* deg, int E)
{
  int e = blockIdx.x * blockDim.x + threadIdx.x;
  if (e >= E) return;
  int s = src[e], d = dst[e];
  if (nmap) { s = nmap[s]; d = nmap[d]; if (s < 0 || d < 0) return; }
  atomicAdd(&deg[d], 1.0f);
}

__global__ void rsqrt_k(const float* deg, float* dis, int n)
{
  int i = blockIdx.x * blockDim.x + threadIdx.x;
  if (i < n) dis[i] = rsqrtf(deg[i]);
}

// out[n, f] = xw[n, f] * dis[n]^2 + bias[f]   (out has row stride ldo, F=128)
__global__ void gcn_init(const float* __restrict__ xw, float* __restrict__ out, int ldo,
                         const float* __restrict__ dis, const float* __restrict__ bias, int Nn)
{
  int idx = blockIdx.x * blockDim.x + threadIdx.x;
  if (idx >= Nn * 128) return;
  int n = idx >> 7, f = idx & 127;
  float d = dis[n];
  out[(long)n * ldo + f] = xw[(long)n * 128 + f] * d * d + bias[f];
}

// out[dst, f] += xw[src, f] * dis[src]*dis[dst]  (wave per edge, 4 feats/lane, F=128)
__global__ void gcn_scatter(const float* __restrict__ xw, float* __restrict__ out, int ldo,
                            const float* __restrict__ dis,
                            const int* __restrict__ src, const int* __restrict__ dst,
                            const int* __restrict__ nmap, int E)
{
  long tid = blockIdx.x * (long)blockDim.x + threadIdx.x;
  long e = tid >> 5;
  if (e >= E) return;
  int f0 = (int)(tid & 31) * 4;
  int s = src[e], d = dst[e];
  if (nmap) { s = nmap[s]; d = nmap[d]; if (s < 0 || d < 0) return; }
  float norm = dis[s] * dis[d];
  const float* xp = xw + (long)s * 128 + f0;
  float* op = out + (long)d * ldo + f0;
  atomicAdd(op + 0, xp[0] * norm);
  atomicAdd(op + 1, xp[1] * norm);
  atomicAdd(op + 2, xp[2] * norm);
  atomicAdd(op + 3, xp[3] * norm);
}

__global__ void relu_strided(float* out, int ldo, int Nn)
{
  int idx = blockIdx.x * blockDim.x + threadIdx.x;
  if (idx >= Nn * 128) return;
  int n = idx >> 7, f = idx & 127;
  float* p = out + (long)n * ldo + f;
  *p = fmaxf(*p, 0.0f);
}

// ---------------------------------------------------------------------------
// Attention pool #1 over xcat (65536 x 384), 128 segments of 512 nodes
// ---------------------------------------------------------------------------
__global__ void seg_reduce_big(const float* __restrict__ xcat,
                               const float* __restrict__ wgt,
                               float* __restrict__ out, float scale)
{
  int s = blockIdx.x;        // 0..127 (graph1 then graph2)
  int f = threadIdx.x;       // 0..383
  int graph = s >> 6, b = s & 63;
  float sum = 0.0f;
  for (int i = 0; i < 512; ++i) {
    int row = b * NPB + graph * 512 + i;
    float w = wgt ? wgt[graph * 32768 + b * 512 + i] : 1.0f;
    sum += xcat[(long)row * NCATQ + f] * w;
  }
  out[s * NCATQ + f] = sum * scale;
}

__global__ void alpha_big(const float* __restrict__ xcat, const float* __restrict__ catt,
                          float* __restrict__ alpha)
{
  int n = blockIdx.x * blockDim.x + threadIdx.x;
  if (n >= NTOT) return;
  int graph = n >> 15, loc = n & 32767;
  int b = loc >> 9, i = loc & 511;
  int row = b * NPB + graph * 512 + i;
  int s = graph * 64 + b;
  const float* xr = xcat + (long)row * NCATQ;
  const float* cr = catt + (long)s * NCATQ;
  float dot = 0.0f;
  for (int f = 0; f < NCATQ; ++f) dot += xr[f] * cr[f];
  alpha[n] = sigmoidf_(dot);
}

__global__ void repack_gpcat(const float* __restrict__ gp, float* __restrict__ gpc)
{
  int t = blockIdx.x * blockDim.x + threadIdx.x;
  if (t >= BQ * 2 * NCATQ) return;
  int b = t / (2 * NCATQ), j = t % (2 * NCATQ);
  gpc[t] = (j < NCATQ) ? gp[b * NCATQ + j] : gp[(b + BQ) * NCATQ + (j - NCATQ)];
}

// ---------------------------------------------------------------------------
// CAG pool: scores, top-k (rank by O(n^2) count -> deterministic), node_map
// ---------------------------------------------------------------------------
__global__ void pvnorm_k(const float* __restrict__ pv, float* __restrict__ pvn)
{
  int t = threadIdx.x;        // 0..127: graph*64 + b
  if (t >= 128) return;
  int graph = t >> 6, b = t & 63;
  const float* p = pv + (long)b * (2 * NCATQ) + graph * NCATQ;
  float s = 0.0f;
  for (int j = 0; j < NCATQ; ++j) s += p[j] * p[j];
  pvn[t] = sqrtf(s);
}

__global__ void score_k(const float* __restrict__ xcat, const float* __restrict__ pv,
                        const float* __restrict__ pvn, float* __restrict__ score)
{
  int n = blockIdx.x * blockDim.x + threadIdx.x;
  if (n >= NTOT) return;
  int graph = n >> 15, loc = n & 32767;
  int b = loc >> 9, i = loc & 511;
  int row = b * NPB + graph * 512 + i;
  const float* xr = xcat + (long)row * NCATQ;
  const float* pr = pv + (long)b * (2 * NCATQ) + graph * NCATQ;
  float dot = 0.0f;
  for (int f = 0; f < NCATQ; ++f) dot += xr[f] * pr[f];
  score[n] = dot / pvn[graph * 64 + b];
}

__global__ void topk_k(const float* __restrict__ score, int* __restrict__ rowmap,
                       float* __restrict__ gate, int* __restrict__ nmap)
{
  int blk = blockIdx.x;          // 0..127
  int graph = blk >> 6, b = blk & 63;
  int i = threadIdx.x;           // 0..511
  __shared__ float s[512];
  const float* sc = score + graph * 32768 + b * 512;
  s[i] = sc[i];
  __syncthreads();
  float my = s[i];
  int rank = 0;
  for (int j = 0; j < 512; ++j) {
    float v = s[j];
    rank += (v > my) || (v == my && j < i);   // argsort(-score) tie-break
  }
  int nm = graph * 32768 + b * 512 + i;
  if (rank < KSEL) {
    int base = graph * (BQ * KSEL) + b * KSEL + rank;
    rowmap[base] = b * NPB + graph * 512 + i;   // xcat row of selected node
    gate[base]   = sigmoidf_(my);
    nmap[nm] = b * KSEL + rank;
  } else {
    nmap[nm] = -1;
  }
}

// ---------------------------------------------------------------------------
// Final attention pool (pooled graphs: 64 segments of 256 rows, F=128)
// ---------------------------------------------------------------------------
__global__ void seg_reduce_small(const float* __restrict__ x, const float* __restrict__ wgt,
                                 float* __restrict__ out, float scale)
{
  int b = blockIdx.x;      // 0..63
  int f = threadIdx.x;     // 0..127
  float sum = 0.0f;
  for (int i = 0; i < KSEL; ++i) {
    float w = wgt ? wgt[b * KSEL + i] : 1.0f;
    sum += x[(long)(b * KSEL + i) * 128 + f] * w;
  }
  out[b * 128 + f] = sum * scale;
}

__global__ void alpha_small(const float* __restrict__ hx, const float* __restrict__ cf,
                            float* __restrict__ alpha)
{
  int n = blockIdx.x * blockDim.x + threadIdx.x;
  if (n >= BQ * KSEL) return;
  int b = n >> 8;
  const float* xr = hx + (long)n * 128;
  const float* cr = cf + (long)b * 128;
  float dot = 0.0f;
  for (int f = 0; f < 128; ++f) dot += xr[f] * cr[f];
  alpha[n] = sigmoidf_(dot);
}

__global__ void repack_predcat(const float* __restrict__ g1, const float* __restrict__ g2,
                               float* __restrict__ pc)
{
  int t = blockIdx.x * blockDim.x + threadIdx.x;
  if (t >= BQ * 256) return;
  int b = t >> 8, j = t & 255;
  pc[t] = (j < 128) ? g1[b * 128 + j] : g2[b * 128 + (j - 128)];
}

__global__ void final_k(const float* __restrict__ pr2, const float* __restrict__ Wl3,
                        const float* __restrict__ bl3, float* __restrict__ out)
{
  int t = threadIdx.x;       // 0..127
  if (t >= 128) return;
  int b = t >> 1, c = t & 1;
  float s = bl3[c];
  for (int j = 0; j < 64; ++j) s += pr2[b * 64 + j] * Wl3[j * 2 + c];
  out[b * 2 + c] = s;
}

// ---------------------------------------------------------------------------
extern "C" void kernel_launch(void* const* d_in, const int* in_sizes, int n_in,
                              void* d_out, int out_size, void* d_ws, size_t ws_size,
                              hipStream_t stream)
{
  (void)in_sizes; (void)n_in; (void)out_size; (void)ws_size;

  const float* x       = (const float*)d_in[0];
  const int*   src_all = (const int*)d_in[1];
  const int*   dst_all = (const int*)d_in[2];
  const int*   src_c1  = (const int*)d_in[3];
  const int*   dst_c1  = (const int*)d_in[4];
  const int*   src_c2  = (const int*)d_in[5];
  const int*   dst_c2  = (const int*)d_in[6];
  const float* W1 = (const float*)d_in[7];   const float* b1 = (const float*)d_in[8];
  const float* W2 = (const float*)d_in[9];   const float* b2 = (const float*)d_in[10];
  const float* W3 = (const float*)d_in[11];  const float* b3 = (const float*)d_in[12];
  const float* Wg_att = (const float*)d_in[13];
  const float* Wal = (const float*)d_in[14]; const float* bal = (const float*)d_in[15];
  const float* Wf  = (const float*)d_in[16]; const float* bf  = (const float*)d_in[17];
  const float* Wg_fin = (const float*)d_in[18];
  const float* Wl1 = (const float*)d_in[19]; const float* bl1 = (const float*)d_in[20];
  const float* Wl2 = (const float*)d_in[21]; const float* bl2 = (const float*)d_in[22];
  const float* Wl3 = (const float*)d_in[23]; const float* bl3 = (const float*)d_in[24];
  float* outp = (float*)d_out;

  // ---- workspace layout (floats) ----
  float* ws    = (float*)d_ws;
  float* xcat  = ws;                               // 65536*384
  float* xw    = xcat + (long)NTOT * NCATQ;        // 65536*128 (reused as xwf)
  float* deg   = xw + (long)NTOT * 128;            // 65536
  float* dis   = deg + NTOT;                       // 65536
  float* meanb = dis + NTOT;                       // 128*384
  float* catt  = meanb + 128 * NCATQ;              // 128*384
  float* alpha = catt + 128 * NCATQ;               // 65536
  float* gp    = alpha + NTOT;                     // 128*384
  float* gpc   = gp + 128 * NCATQ;                 // 64*768
  float* pv    = gpc + BQ * 2 * NCATQ;             // 64*768
  float* pvn   = pv + BQ * 2 * NCATQ;              // 128
  float* score = pvn + 128;                        // 65536
  float* gate  = score + NTOT;                     // 32768
  float* degf  = gate + 2 * BQ * KSEL;             // 32768
  float* disf  = degf + 32768;                     // 32768
  float* hf1   = disf + 32768;                     // 16384*128
  float* hf2   = hf1 + (long)BQ * KSEL * 128;      // 16384*128
  float* meanf = hf2 + (long)BQ * KSEL * 128;      // 64*128
  float* cfb   = meanf + BQ * 128;                 // 64*128
  float* alphaf= cfb + BQ * 128;                   // 16384
  float* g1    = alphaf + BQ * KSEL;               // 64*128
  float* g2    = g1 + BQ * 128;                    // 64*128
  float* pcat  = g2 + BQ * 128;                    // 64*256
  float* pr1   = pcat + BQ * 256;                  // 64*128
  float* pr2   = pr1 + BQ * 128;                   // 64*64
  int*   rowmap = (int*)(pr2 + BQ * 64);           // 32768 ints
  int*   nmap   = rowmap + 2 * BQ * KSEL;          // 65536 ints

  auto gemm = [&](const float* A, int lda, const float* W, int ldw,
                  float* C, int ldc, const float* bias,
                  const int* rm, const float* rs, int M, int K, int Nc, int act) {
    int waves = (M / 16) * (Nc / 16);
    int blocks = (waves * 32 + 255) / 256;
    gemm_wmma<<<blocks, 256, 0, stream>>>(A, lda, W, ldw, C, ldc, bias, rm, rs, M, K, Nc, act);
  };

  // ---- degrees & inv-sqrt for the full graph ----
  fill_f32<<<(NTOT + 255) / 256, 256, 0, stream>>>(deg, 1.0f, NTOT);
  deg_accum<<<(E_ALL + 255) / 256, 256, 0, stream>>>(src_all, dst_all, nullptr, deg, E_ALL);
  rsqrt_k<<<(NTOT + 255) / 256, 256, 0, stream>>>(deg, dis, NTOT);

  // ---- 3 GCN layers; layer l writes relu output into xcat column slice l*128 ----
  const float* convA[3] = { x, xcat + 0, xcat + 128 };
  const int    convLda[3] = { NFQ, NCATQ, NCATQ };
  const float* convW[3] = { W1, W2, W3 };
  const float* convB[3] = { b1, b2, b3 };
  for (int l = 0; l < 3; ++l) {
    float* slice = xcat + l * 128;
    gemm(convA[l], convLda[l], convW[l], NHQ, xw, 128, nullptr, nullptr, nullptr,
         NTOT, 128, NHQ, ACT_NONE);
    gcn_init<<<(NTOT * 128 + 255) / 256, 256, 0, stream>>>(xw, slice, NCATQ, dis, convB[l], NTOT);
    gcn_scatter<<<(int)(((long)E_ALL * 32 + 255) / 256), 256, 0, stream>>>(
        xw, slice, NCATQ, dis, src_all, dst_all, nullptr, E_ALL);
    relu_strided<<<(NTOT * 128 + 255) / 256, 256, 0, stream>>>(slice, NCATQ, NTOT);
  }

  // ---- attention pool over xcat (2B segments) ----
  seg_reduce_big<<<128, NCATQ, 0, stream>>>(xcat, nullptr, meanb, 1.0f / 512.0f);
  gemm(meanb, NCATQ, Wg_att, NCATQ, catt, NCATQ, nullptr, nullptr, nullptr,
       128, NCATQ, NCATQ, ACT_TANH);
  alpha_big<<<(NTOT + 255) / 256, 256, 0, stream>>>(xcat, catt, alpha);
  seg_reduce_big<<<128, NCATQ, 0, stream>>>(xcat, alpha, gp, 1.0f);

  // ---- pv = concat(gp[:B], gp[B:]) @ Wal + bal ----
  repack_gpcat<<<(BQ * 2 * NCATQ + 255) / 256, 256, 0, stream>>>(gp, gpc);
  gemm(gpc, 2 * NCATQ, Wal, 2 * NCATQ, pv, 2 * NCATQ, bal, nullptr, nullptr,
       BQ, 2 * NCATQ, 2 * NCATQ, ACT_NONE);

  // ---- CAG pool: scores + top-k selection for both subgraphs ----
  pvnorm_k<<<1, 128, 0, stream>>>(pv, pvn);
  score_k<<<(NTOT + 255) / 256, 256, 0, stream>>>(xcat, pv, pvn, score);
  topk_k<<<128, 512, 0, stream>>>(score, rowmap, gate, nmap);

  // ---- pooled-graph degrees ----
  fill_f32<<<(32768 + 255) / 256, 256, 0, stream>>>(degf, 1.0f, 32768);
  deg_accum<<<(E_C + 255) / 256, 256, 0, stream>>>(src_c1, dst_c1, nmap, degf, E_C);
  deg_accum<<<(E_C + 255) / 256, 256, 0, stream>>>(src_c2, dst_c2, nmap + 32768, degf + 16384, E_C);
  rsqrt_k<<<(32768 + 255) / 256, 256, 0, stream>>>(degf, disf, 32768);

  // ---- masked GCN over each pooled graph (gather+gate fused into GEMM A-load) ----
  const int*   pSrc[2]  = { src_c1, src_c2 };
  const int*   pDst[2]  = { dst_c1, dst_c2 };
  float*       pH[2]    = { hf1, hf2 };
  float*       pG[2]    = { g1, g2 };
  const int MP = BQ * KSEL;   // 16384
  for (int g = 0; g < 2; ++g) {
    const int*   rm  = rowmap + g * MP;
    const float* rs  = gate + g * MP;
    const int*   nm  = nmap + g * 32768;
    const float* dsf = disf + g * MP;
    gemm(xcat, NCATQ, Wf, NHQ, xw, 128, nullptr, rm, rs, MP, NCATQ, NHQ, ACT_NONE);
    gcn_init<<<(MP * 128 + 255) / 256, 256, 0, stream>>>(xw, pH[g], 128, dsf, bf, MP);
    gcn_scatter<<<(int)(((long)E_C * 32 + 255) / 256), 256, 0, stream>>>(
        xw, pH[g], 128, dsf, pSrc[g], pDst[g], nm, E_C);
    relu_strided<<<(MP * 128 + 255) / 256, 256, 0, stream>>>(pH[g], 128, MP);

    // final attention pool
    seg_reduce_small<<<BQ, 128, 0, stream>>>(pH[g], nullptr, meanf, 1.0f / (float)KSEL);
    gemm(meanf, 128, Wg_fin, NHQ, cfb, 128, nullptr, nullptr, nullptr, BQ, NHQ, NHQ, ACT_TANH);
    alpha_small<<<(MP + 255) / 256, 256, 0, stream>>>(pH[g], cfb, alphaf);
    seg_reduce_small<<<BQ, 128, 0, stream>>>(pH[g], alphaf, pG[g], 1.0f);
  }

  // ---- MLP head ----
  repack_predcat<<<(BQ * 256 + 255) / 256, 256, 0, stream>>>(g1, g2, pcat);
  gemm(pcat, 256, Wl1, NHQ, pr1, 128, bl1, nullptr, nullptr, BQ, 2 * NHQ, NHQ, ACT_RELU);
  gemm(pr1, 128, Wl2, NHQ / 2, pr2, 64, bl2, nullptr, nullptr, BQ, NHQ, NHQ / 2, ACT_RELU);
  final_k<<<1, 128, 0, stream>>>(pr2, Wl3, bl3, outp);
}
